// WindowAttention3D_S_9964324126896
// MI455X (gfx1250) — compile-verified, matrix operand order checked by reading, not run on hardware
//
#include <hip/hip_runtime.h>
#include <hip/hip_bf16.h>

typedef __bf16 bf16_t;
typedef bf16_t v16bf __attribute__((ext_vector_type(16)));
typedef float  v8f   __attribute__((ext_vector_type(8)));

#define DIM   128
#define HEADS 4
#define HD    32
#define NTOK  128
#define QK_SCALE 0.17677669529663687f   // 32^-0.5

// LDS row strides (bf16 elements); all rows stay 16-byte aligned
#define LDX   136
#define LDQ   40
#define LDV   136
#define LDSS  136
#define LDW   136
#define LDO   136

// flat LDS pool layout (byte offsets, all 16B aligned)
#define OFF_X   0u        // Xs  bf16 [128][136]               34816 B
#define OFF_Q   34816u    // Qs  bf16 [128][40]                10240 B
#define OFF_K   45056u    // Ks  bf16 [128][40]                10240 B
#define OFF_V   55296u    // Vs  bf16 [32][136] (transposed)    8704 B
#define OFF_SW  64000u    // Ss logits / Ws qkv-slice / Wp proj 34816 B
#define OFF_O   98816u    // Os  bf16 [128][136]               34816 B
#define POOL_BYTES 133632u
// transient f32 X staging aliases [OFF_SW, OFF_SW+65536) = dead at kernel start

__device__ __forceinline__ v8f wmma_bf16(v16bf a, v16bf b, v8f c) {
  return __builtin_amdgcn_wmma_f32_16x16x32_bf16(false, a, false, b, (short)0, c, false, false);
}

// fragment from LDS bf16 row: two 16B chunks at K offsets start, start+16
__device__ __forceinline__ v16bf frag_lds(const bf16_t* rowp, int start) {
  union { uint4 u[2]; v16bf v; } f;
  f.u[0] = *(const uint4*)(rowp + start);
  f.u[1] = *(const uint4*)(rowp + start + 16);
  return f.v;
}

__device__ __forceinline__ void store4_bf16(bf16_t* dst, float a, float b,
                                            float c, float d) {
  union { bf16_t h[4]; uint2 u; } p;
  p.h[0] = (bf16_t)a; p.h[1] = (bf16_t)b; p.h[2] = (bf16_t)c; p.h[3] = (bf16_t)d;
  *(uint2*)dst = p.u;
}

__global__ __launch_bounds__(256)
void win_attn3d_kernel(const float* __restrict__ x,
                       const int*   __restrict__ rpi,
                       const float* __restrict__ maskw,
                       const float* __restrict__ sp_mask,
                       const float* __restrict__ w_qkv,
                       const float* __restrict__ b_qkv,
                       const float* __restrict__ bias_table,
                       const float* __restrict__ w_proj,
                       const float* __restrict__ b_proj,
                       float* __restrict__ out) {
  __shared__ __align__(16) unsigned char pool[POOL_BYTES];
  bf16_t* Xs = (bf16_t*)(pool + OFF_X);
  bf16_t* Qs = (bf16_t*)(pool + OFF_Q);
  bf16_t* Ks = (bf16_t*)(pool + OFF_K);
  bf16_t* Vs = (bf16_t*)(pool + OFF_V);
  bf16_t* SW = (bf16_t*)(pool + OFF_SW);   // Ss / Ws / Wp (time-multiplexed)
  bf16_t* Os = (bf16_t*)(pool + OFF_O);
  float*  Xf = (float*)(pool + OFF_SW);    // transient raw-f32 X

  const int b    = blockIdx.x;
  const int tid  = threadIdx.x;
  const int lane = tid & 31;
  const int wv   = tid >> 5;        // 8 waves
  const int half = lane >> 4;
  const int l16  = lane & 15;

  // Warm L2 with the (block-shared) weights
  for (int i = tid; i < (3 * DIM * DIM) / 64; i += 256)
    __builtin_prefetch(w_qkv + i * 64, 0, 0);
  for (int i = tid; i < (DIM * DIM) / 64; i += 256)
    __builtin_prefetch(w_proj + i * 64, 0, 0);

  // ---- async-stage X[b] (raw f32) into transient LDS region ----
  const float* xb = x + (size_t)b * NTOK * DIM;
  for (int i = tid * 4; i < NTOK * DIM; i += 256 * 4) {
    unsigned dst = OFF_SW + (unsigned)i * 4u;     // LDS byte offset
    const float* g = xb + i;
    asm volatile("global_load_async_to_lds_b128 %0, %1, off"
                 :: "v"(dst), "v"(g) : "memory");
  }
  asm volatile("s_wait_asynccnt 0x0" ::: "memory");
  __syncthreads();

  // ---- convert f32 -> bf16 Xs (padded stride) ----
  for (int i = tid * 4; i < NTOK * DIM; i += 256 * 4) {
    float4 f = *(const float4*)(Xf + i);
    store4_bf16(&Xs[(i >> 7) * LDX + (i & 127)], f.x, f.y, f.z, f.w);
  }
  __syncthreads();

  // hoisted X A-fragments (identical for every head)
  v16bf afr[4];
  #pragma unroll
  for (int kk = 0; kk < 4; ++kk)
    afr[kk] = frag_lds(Xs + (wv * 16 + l16) * LDX, kk * 32 + half * 8);

  const float* spb = sp_mask + (size_t)(b & 255) * NTOK * NTOK;
  const float* mkb = maskw   + (size_t)(b & 255) * NTOK * NTOK;

  for (int h = 0; h < HEADS; ++h) {
    // ---- stage per-head QKV weight slice (96 rows x 128) into SW as bf16 ----
    for (int i = tid * 4; i < 96 * 128; i += 256 * 4) {
      const int rloc = i >> 7, c = i & 127;
      const int sec = rloc >> 5, dd = rloc & 31;
      float4 f = *(const float4*)(w_qkv + (size_t)(sec * DIM + h * HD + dd) * DIM + c);
      store4_bf16(&SW[rloc * LDW + c], f.x, f.y, f.z, f.w);
    }
    __syncthreads();

    // ---- GEMM1: Qh/Kh/Vh = X * Wslice^T (+bias), dual accumulators ----
    {
      const int mt = wv;
      #pragma unroll
      for (int sec = 0; sec < 3; ++sec) {
        const bf16_t* w0 = SW + (sec * 32 + l16) * LDW;
        const bf16_t* w1 = SW + (sec * 32 + 16 + l16) * LDW;
        v8f acc0 = {}, acc1 = {};
        #pragma unroll
        for (int kk = 0; kk < 4; ++kk) {
          const int st = kk * 32 + half * 8;
          acc0 = wmma_bf16(afr[kk], frag_lds(w0, st), acc0);
          acc1 = wmma_bf16(afr[kk], frag_lds(w1, st), acc1);
        }
        const int   o0 = sec * DIM + h * HD + l16;
        const float bias0 = b_qkv[o0], bias1 = b_qkv[o0 + 16];
        const int   d0 = l16, d1 = 16 + l16;
        #pragma unroll
        for (int r = 0; r < 8; ++r) {
          const int row = mt * 16 + half * 8 + r;
          const float v0 = acc0[r] + bias0, v1 = acc1[r] + bias1;
          if (sec == 0) {
            Qs[row * LDQ + d0] = (bf16_t)(v0 * QK_SCALE);
            Qs[row * LDQ + d1] = (bf16_t)(v1 * QK_SCALE);
          } else if (sec == 1) {
            Ks[row * LDQ + d0] = (bf16_t)v0;
            Ks[row * LDQ + d1] = (bf16_t)v1;
          } else {
            Vs[d0 * LDV + row] = (bf16_t)v0;    // transposed
            Vs[d1 * LDV + row] = (bf16_t)v1;
          }
        }
      }
    }
    __syncthreads();

    // ---- GEMM2: S = Qh * Kh^T (K=32), dual accumulators ----
    {
      const int mt = wv;
      const v16bf qf = frag_lds(Qs + (mt * 16 + l16) * LDQ, half * 8);
      #pragma unroll
      for (int np = 0; np < 4; ++np) {
        v8f a0 = {}, a1 = {};
        a0 = wmma_bf16(qf, frag_lds(Ks + (np * 32 + l16) * LDQ, half * 8), a0);
        a1 = wmma_bf16(qf, frag_lds(Ks + (np * 32 + 16 + l16) * LDQ, half * 8), a1);
        #pragma unroll
        for (int r = 0; r < 8; ++r) {
          const int row = mt * 16 + half * 8 + r;
          SW[row * LDSS + np * 32 + l16]      = (bf16_t)a0[r];
          SW[row * LDSS + np * 32 + 16 + l16] = (bf16_t)a1[r];
        }
      }
    }
    __syncthreads();

    // ---- bias add + row softmax (2 threads/row, packed 4-wide) ----
    {
      const int row = tid >> 1;
      const int c0  = (tid & 1) * 64;
      const int*   rp = rpi + row * NTOK;
      const float* sp = spb + (size_t)row * NTOK;
      const float* mk = mkb + (size_t)row * NTOK;
      union P4 { uint2 u; bf16_t hh[4]; };
      float mx = -1e30f;
      for (int j = 0; j < 64; j += 4) {
        const int c = c0 + j;
        P4 s; s.u = *(uint2*)&SW[row * LDSS + c];
        const int4   ri  = *(const int4*)(rp + c);
        const float4 spv = *(const float4*)(sp + c);
        const float4 mkv = *(const float4*)(mk + c);
        const float v0 = (float)s.hh[0] + bias_table[ri.x * HEADS + h] + spv.x + mkv.x;
        const float v1 = (float)s.hh[1] + bias_table[ri.y * HEADS + h] + spv.y + mkv.y;
        const float v2 = (float)s.hh[2] + bias_table[ri.z * HEADS + h] + spv.z + mkv.z;
        const float v3 = (float)s.hh[3] + bias_table[ri.w * HEADS + h] + spv.w + mkv.w;
        store4_bf16(&SW[row * LDSS + c], v0, v1, v2, v3);
        mx = fmaxf(mx, fmaxf(fmaxf(v0, v1), fmaxf(v2, v3)));
      }
      mx = fmaxf(mx, __shfl_xor(mx, 1, 32));
      float sum = 0.f;
      for (int j = 0; j < 64; j += 4) {
        const int c = c0 + j;
        P4 s; s.u = *(uint2*)&SW[row * LDSS + c];
        const float e0 = __expf((float)s.hh[0] - mx);
        const float e1 = __expf((float)s.hh[1] - mx);
        const float e2 = __expf((float)s.hh[2] - mx);
        const float e3 = __expf((float)s.hh[3] - mx);
        store4_bf16(&SW[row * LDSS + c], e0, e1, e2, e3);
        sum += (e0 + e1) + (e2 + e3);
      }
      sum += __shfl_xor(sum, 1, 32);
      const float inv = 1.0f / sum;
      for (int j = 0; j < 64; j += 4) {
        const int c = c0 + j;
        P4 s; s.u = *(uint2*)&SW[row * LDSS + c];
        store4_bf16(&SW[row * LDSS + c], (float)s.hh[0] * inv, (float)s.hh[1] * inv,
                    (float)s.hh[2] * inv, (float)s.hh[3] * inv);
      }
    }
    __syncthreads();

    // ---- GEMM3: Oh = P * Vh (N=32, K=128), dual accumulators ----
    {
      const int mt = wv;
      v16bf pf[4];
      #pragma unroll
      for (int kk = 0; kk < 4; ++kk)
        pf[kk] = frag_lds(SW + (mt * 16 + l16) * LDSS, kk * 32 + half * 8);
      v8f a0 = {}, a1 = {};
      #pragma unroll
      for (int kk = 0; kk < 4; ++kk) {
        const int st = kk * 32 + half * 8;
        a0 = wmma_bf16(pf[kk], frag_lds(Vs + l16 * LDV, st), a0);
        a1 = wmma_bf16(pf[kk], frag_lds(Vs + (16 + l16) * LDV, st), a1);
      }
      #pragma unroll
      for (int r = 0; r < 8; ++r) {
        const int row = mt * 16 + half * 8 + r;
        Os[row * LDO + h * HD + l16]      = (bf16_t)a0[r];
        Os[row * LDO + h * HD + 16 + l16] = (bf16_t)a1[r];
      }
    }
    __syncthreads();   // protect Qs/Ks/Vs/SW before next head rewrites them
  }

  // ---- stage w_proj (128x128) into SW as bf16 ----
  for (int i = tid * 4; i < DIM * DIM; i += 256 * 4) {
    const int rloc = i >> 7, c = i & 127;
    float4 f = *(const float4*)(w_proj + (size_t)rloc * DIM + c);
    store4_bf16(&SW[rloc * LDW + c], f.x, f.y, f.z, f.w);
  }
  __syncthreads();

  // ---- GEMM4: Y = O * Wproj^T + b_proj -> global f32, dual accumulators ----
  {
    const int mt = wv;
    v16bf ofr[4];
    #pragma unroll
    for (int kk = 0; kk < 4; ++kk)
      ofr[kk] = frag_lds(Os + (mt * 16 + l16) * LDO, kk * 32 + half * 8);
    float* yb = out + (size_t)b * NTOK * DIM;
    #pragma unroll
    for (int np = 0; np < 4; ++np) {
      const int n0 = np * 32 + l16, n1 = np * 32 + 16 + l16;
      const bf16_t* w0 = SW + n0 * LDW;
      const bf16_t* w1 = SW + n1 * LDW;
      v8f a0 = {}, a1 = {};
      #pragma unroll
      for (int kk = 0; kk < 4; ++kk) {
        const int st = kk * 32 + half * 8;
        a0 = wmma_bf16(ofr[kk], frag_lds(w0, st), a0);
        a1 = wmma_bf16(ofr[kk], frag_lds(w1, st), a1);
      }
      const float bias0 = b_proj[n0], bias1 = b_proj[n1];
      #pragma unroll
      for (int r = 0; r < 8; ++r) {
        const int row = mt * 16 + half * 8 + r;
        yb[row * DIM + n0] = a0[r] + bias0;
        yb[row * DIM + n1] = a1[r] + bias1;
      }
    }
  }
}

extern "C" void kernel_launch(void* const* d_in, const int* in_sizes, int n_in,
                              void* d_out, int out_size, void* d_ws, size_t ws_size,
                              hipStream_t stream) {
  const float* x        = (const float*)d_in[0];
  const int*   rpi      = (const int*)  d_in[1];
  const float* maskw    = (const float*)d_in[2];
  const float* sp_mask  = (const float*)d_in[3];
  const float* w_qkv    = (const float*)d_in[4];
  const float* b_qkv    = (const float*)d_in[5];
  const float* btab     = (const float*)d_in[6];
  const float* w_proj   = (const float*)d_in[7];
  const float* b_proj   = (const float*)d_in[8];
  float* out = (float*)d_out;

  dim3 grid(2048), block(256);
  hipLaunchKernelGGL(win_attn3d_kernel, grid, block, 0, stream,
                     x, rpi, maskw, sp_mask, w_qkv, b_qkv, btab, w_proj, b_proj, out);
}
